// GATLayer_19370302505052
// MI455X (gfx1250) — matrix operand order repaired and checked by reading, not compiled, not run on hardware
//
#include <hip/hip_runtime.h>
#include <hip/hip_bf16.h>

typedef __attribute__((ext_vector_type(16))) _Float16 v16h;
typedef __attribute__((ext_vector_type(8)))  float    v8f;

union AFrag { v16h v; _Float16 e[16]; };
union CFrag { v8f  v; float     e[8];  };

#define F_IN  128
#define DIM   32
#define HEADS 4
#define OUTW  128          // DIM*HEADS
#define PACKW_ELEMS (2*8*4*32*16)   // sel x ntile x kb x lane x 16 = 32768 f16

// --------------------------------------------------------------------------
// Kernel 0: pre-pack W_l/W_r (fp32, row-major 128x128) into fragment-ready
// f16 B-operand layout:  packed[((sel*8+ntile)*4+kb)*32 + lane][0..15]
// where element i of lane holds W[kb*32 + (lane>>4)*16 + i][ntile*16 + (lane&15)]
// (B 32x16 in this code's k order, A matches: lane%16 = N, lane/16 picks K half, elems = K offset)
// --------------------------------------------------------------------------
__global__ void gat_pack_w(const float* __restrict__ Wl,
                           const float* __restrict__ Wr,
                           _Float16* __restrict__ packed)
{
  const int i = blockIdx.x * blockDim.x + threadIdx.x;
  if (i >= PACKW_ELEMS) return;
  const int elem  = i & 15;
  const int lane  = (i >> 4) & 31;
  const int kb    = (i >> 9) & 3;
  const int ntile = (i >> 11) & 7;
  const int sel   = (i >> 14) & 1;
  const float* W = sel ? Wr : Wl;
  const int k   = kb * 32 + (lane >> 4) * 16 + elem;
  const int col = ntile * 16 + (lane & 15);
  packed[i] = (_Float16)W[k * OUTW + col];
}

// --------------------------------------------------------------------------
// Kernel 1: fused dual projection via WMMA, B operands staged in LDS.
// One wave per 16-row tile; 4-wave workgroup shares the 64KB packed-W table.
// --------------------------------------------------------------------------
__global__ __launch_bounds__(128) void gat_project_wmma(
    const float* __restrict__ nodes,
    const _Float16* __restrict__ packedW,
    const float* __restrict__ bl, const float* __restrict__ br,
    float* __restrict__ ql, float* __restrict__ qr, int N)
{
  __shared__ _Float16 ldsW[PACKW_ELEMS];   // 64 KB

  // Cooperative stage: 4096 uint4 (16B) chunks across 128 threads
  {
    const uint4* src = (const uint4*)packedW;
    uint4*       dst = (uint4*)ldsW;
    #pragma unroll
    for (int i = 0; i < PACKW_ELEMS / 8 / 128; ++i)
      dst[threadIdx.x + i * 128] = src[threadIdx.x + i * 128];
  }
  __syncthreads();

  const int lane   = threadIdx.x & 31;
  const int wv     = threadIdx.x >> 5;
  const int mtile  = blockIdx.x * 4 + wv;
  const int mtiles = (N + 15) >> 4;
  if (mtile >= mtiles) return;

  const int m = lane & 15;
  const int h = lane >> 4;
  int row = mtile * 16 + m;
  if (row >= N) row = N - 1;            // clamp loads; stores are guarded

  // Wave-uniform: does this tile write a full 16 rows? (single scalar branch)
  const bool full = (mtile * 16 + 16) <= N;

  // Preload + convert A fragments for the full K=128 (4 fragments of K=32)
  // A (16-bit 16x32): lane&15 = M, lane>>4 = h:
  //   elems 0..15 -> K = h*16+0..15
  AFrag a[4];
  const float* arow = nodes + (size_t)row * F_IN;
  #pragma unroll
  for (int kb = 0; kb < 4; ++kb) {
    const float* ap = arow + kb * 32 + h * 16;
    #pragma unroll
    for (int i = 0; i < 8; ++i) {
      a[kb].e[i]     = (_Float16)ap[i];
      a[kb].e[8 + i] = (_Float16)ap[8 + i];
    }
  }

  const int rbase = mtile * 16 + h * 8;

  #pragma unroll
  for (int t = 0; t < 16; ++t) {
    const int sel   = t >> 3;           // 0 -> left proj, 1 -> right proj
    const int ntile = t & 7;
    const float* bias = sel ? br : bl;
    float*       outp = sel ? qr : ql;
    const int col0 = ntile * 16;

    CFrag acc;
    #pragma unroll
    for (int r = 0; r < 8; ++r) acc.e[r] = 0.0f;

    #pragma unroll
    for (int kb = 0; kb < 4; ++kb) {
      const _Float16* bp = ldsW + (((size_t)(t * 4 + kb) * 32 + lane) << 4);
      const v16h bv = *(const v16h*)bp;  // 32B contiguous -> ds_load_b128 x2
      acc.v = __builtin_amdgcn_wmma_f32_16x16x32_f16(
          false, a[kb].v, false, bv, (short)0, acc.v, false, false);
    }

    // C (f32 16x16): vgpr r -> M = h*8 + r, N = lane&15
    const float bv = bias[col0 + m];
    float* op = outp + (size_t)rbase * OUTW + col0 + m;
    if (full) {
      #pragma unroll
      for (int r = 0; r < 8; ++r)        // constant offsets fold into IOFFSET
        op[r * OUTW] = acc.e[r] + bv;
    } else {
      #pragma unroll
      for (int r = 0; r < 8; ++r)
        if (rbase + r < N) op[r * OUTW] = acc.e[r] + bv;
    }
  }
}

// --------------------------------------------------------------------------
// Order-preserving uint encoding of float (for atomicMax-based segment max)
// --------------------------------------------------------------------------
__device__ __forceinline__ unsigned enc_f32(float f) {
  unsigned u = __float_as_uint(f);
  return (u & 0x80000000u) ? ~u : (u | 0x80000000u);
}
__device__ __forceinline__ float dec_f32(unsigned u) {
  return (u & 0x80000000u) ? __uint_as_float(u & 0x7fffffffu)
                           : __uint_as_float(~u);
}

// Kernel 2: init encoded maxs (0 == below any real float), denom, out
__global__ void gat_init(unsigned* __restrict__ maxs,
                         float* __restrict__ denom,
                         float* __restrict__ out, int N)
{
  const int i = blockIdx.x * blockDim.x + threadIdx.x;
  if (i < N * HEADS) { maxs[i] = 0u; denom[i] = 0.0f; }
  if (i < N * OUTW)  out[i] = 0.0f;
}

// --------------------------------------------------------------------------
// Kernel 3: per-edge attention logits + segment max.
// One wave per edge; lane = feature d in [0,32). Wave32 shuffle reduction.
// --------------------------------------------------------------------------
__global__ __launch_bounds__(256) void gat_edge_logits(
    const float* __restrict__ ql, const float* __restrict__ qr,
    const int* __restrict__ senders, const int* __restrict__ receivers,
    const float* __restrict__ w_a, const float* __restrict__ b_a,
    float* __restrict__ logits, unsigned* __restrict__ maxs, int E)
{
  const int lane = threadIdx.x & 31;
  const int e    = blockIdx.x * 8 + (threadIdx.x >> 5);
  if (e >= E) return;
  const int s = senders[e];
  const int r = receivers[e];
  const float wa = w_a[lane];
  const float ba = b_a[0];
  #pragma unroll
  for (int hh = 0; hh < HEADS; ++hh) {
    const float x = ql[(size_t)s * OUTW + hh * DIM + lane] +
                    qr[(size_t)r * OUTW + hh * DIM + lane];
    float p = ((x > 0.0f) ? x : 0.2f * x) * wa;   // leaky_relu(0.2) * w_a
    #pragma unroll
    for (int off = 16; off > 0; off >>= 1) p += __shfl_xor(p, off, 32);
    if (lane == 0) {
      const float lg = p + ba;
      logits[(size_t)e * HEADS + hh] = lg;
      atomicMax(&maxs[r * HEADS + hh], enc_f32(lg));
    }
  }
}

// Kernel 4: unnorm = exp(logit - max) (overwrites logits), denom += unnorm
__global__ void gat_edge_expsum(
    float* __restrict__ logits, const unsigned* __restrict__ maxs,
    const int* __restrict__ receivers, float* __restrict__ denom, int E4)
{
  const int i = blockIdx.x * blockDim.x + threadIdx.x;
  if (i >= E4) return;
  const int e  = i >> 2;
  const int hh = i & 3;
  const int r  = receivers[e];
  const float mx = dec_f32(maxs[r * HEADS + hh]);
  const float w  = __expf(logits[i] - mx);
  logits[i] = w;
  atomicAdd(&denom[r * HEADS + hh], w);
}

// Kernel 5: out[r] += (unnorm/denom[r]) * sent ; wave per edge, lane = d
__global__ __launch_bounds__(256) void gat_edge_scatter(
    const float* __restrict__ ql, const float* __restrict__ unnorm,
    const float* __restrict__ denom,
    const int* __restrict__ senders, const int* __restrict__ receivers,
    float* __restrict__ out, int E)
{
  const int lane = threadIdx.x & 31;
  const int e    = blockIdx.x * 8 + (threadIdx.x >> 5);
  if (e >= E) return;
  const int s = senders[e];
  const int r = receivers[e];
  #pragma unroll
  for (int hh = 0; hh < HEADS; ++hh) {
    const float alpha = unnorm[(size_t)e * HEADS + hh] /
                        denom[r * HEADS + hh];
    const float v = alpha * ql[(size_t)s * OUTW + hh * DIM + lane];
    atomicAdd(&out[(size_t)r * OUTW + hh * DIM + lane], v);
  }
}

// --------------------------------------------------------------------------
extern "C" void kernel_launch(void* const* d_in, const int* in_sizes, int n_in,
                              void* d_out, int out_size, void* d_ws, size_t ws_size,
                              hipStream_t stream)
{
  const float* nodes     = (const float*)d_in[0];
  const int*   senders   = (const int*)  d_in[1];
  const int*   receivers = (const int*)  d_in[2];
  const float* W_l       = (const float*)d_in[3];
  const float* b_l       = (const float*)d_in[4];
  const float* W_r       = (const float*)d_in[5];
  const float* b_r       = (const float*)d_in[6];
  const float* w_a       = (const float*)d_in[7];
  const float* b_a       = (const float*)d_in[8];

  const int N = in_sizes[0] / F_IN;
  const int E = in_sizes[1];

  float* out = (float*)d_out;
  float* ws  = (float*)d_ws;

  // Workspace layout (packedW first so it keeps 16B alignment):
  _Float16* packedW = (_Float16*)ws;                       // 64 KB
  float*    ql      = ws + PACKW_ELEMS / 2;                // N*128 f32
  float*    qr      = ql + (size_t)N * OUTW;               // N*128 f32
  float*    logits  = qr + (size_t)N * OUTW;               // E*4  f32
  unsigned* maxs    = (unsigned*)(logits + (size_t)E * HEADS);  // N*4
  float*    denom   = (float*)(maxs + (size_t)N * HEADS);       // N*4

  gat_pack_w<<<(PACKW_ELEMS + 255) / 256, 256, 0, stream>>>(W_l, W_r, packedW);

  const int mtiles = (N + 15) / 16;
  gat_project_wmma<<<(mtiles + 3) / 4, 128, 0, stream>>>(
      nodes, packedW, b_l, b_r, ql, qr, N);

  const int initn = N * OUTW;
  gat_init<<<(initn + 255) / 256, 256, 0, stream>>>(maxs, denom, out, N);

  gat_edge_logits<<<(E + 7) / 8, 256, 0, stream>>>(
      ql, qr, senders, receivers, w_a, b_a, logits, maxs, E);

  const int E4 = E * HEADS;
  gat_edge_expsum<<<(E4 + 255) / 256, 256, 0, stream>>>(
      logits, maxs, receivers, denom, E4);

  gat_edge_scatter<<<(E + 7) / 8, 256, 0, stream>>>(
      ql, logits, denom, senders, receivers, out, E);
}
